// ResAttention_40707700032414
// MI455X (gfx1250) — matrix operand order repaired and checked by reading, not compiled, not run on hardware
//
#include <hip/hip_runtime.h>
#include <math.h>

// MI455X (gfx1250) attention forward, fp32 in/out, f16 WMMA compute (f32 acc).
// Two-pass flash-style: (1) scores + per-lane online softmax stats,
// (2) normalize + P@V. HBM-bound on the 1 GiB attention map (~3.2 GiB total),
// so attn traffic is non-temporal to keep the reusable 32 MB K/V in L2.

typedef __attribute__((ext_vector_type(16))) _Float16 v16h;
typedef __attribute__((ext_vector_type(8)))  float    v8f;

constexpr int Bn = 4, Ln = 2048, Sn = 2048, Hn = 16, En = 64;
constexpr float SCALE = 0.125f; // 1/sqrt(64), folded into Q->f16 conversion
constexpr int WAVES_PER_BLOCK = 4;

// ---------------------------------------------------------------------------
// Pass 1: scores = (Q*scale) K^T, stored raw (non-temporal) into attn buffer.
// Online softmax stats kept PER LANE (each lane owns one column of every
// tile), merged across the 16 lanes of each half only once at the end.
// One wave handles a 16-row query block for one (b,h); S-step = 32.
// ---------------------------------------------------------------------------
__global__ __launch_bounds__(128) void attn_scores_kernel(
    const float* __restrict__ q, const float* __restrict__ k,
    float* __restrict__ attn, float* __restrict__ mbuf, float* __restrict__ lbuf)
{
    const int lane = threadIdx.x & 31;
    const int wave = threadIdx.x >> 5;
    const int wid  = blockIdx.x * WAVES_PER_BLOCK + wave;

    const int lblocks = Ln / 16;
    const int lb  = wid % lblocks;
    int tmp       = wid / lblocks;
    const int h   = tmp % Hn;
    const int b   = tmp / Hn;
    const int l0  = lb * 16;

    const int ln   = lane & 15;   // A: row M, B: col N, C: col N
    const int half = lane >> 4;   // lane half

    // --- A fragment (Q rows, 16x32 f16 layout), scale folded in, reused ---
    // element i of v16h -> K index: i+16*half (contiguous, matches B)
    const float* qrow = q + ((size_t)(b * Ln + l0 + ln) * Hn + h) * En;
    v16h a0, a1;
#pragma unroll
    for (int i = 0; i < 16; ++i) {
        const int e = i + 16 * half;
        a0[i] = (_Float16)(qrow[e] * SCALE);
        a1[i] = (_Float16)(qrow[e + 32] * SCALE);
    }

    // per-lane partial softmax state for rows M = r + 8*half
    float mpart[8], lpart[8];
#pragma unroll
    for (int r = 0; r < 8; ++r) { mpart[r] = -__builtin_inff(); lpart[r] = 0.f; }

    float* attn_base = attn + ((size_t)(b * Hn + h) * Ln + l0) * Sn;

    for (int sc0 = 0; sc0 < Sn; sc0 += 32) {
#pragma unroll
        for (int t = 0; t < 2; ++t) {
            const int s0 = sc0 + t * 16;
            // B fragment (K^T, 32x16): lane covers key row s0+ln, 16 consec e.
            const float* krow = k + ((size_t)(b * Sn + s0 + ln) * Hn + h) * En + 16 * half;
            if (s0 + 16 < Sn) __builtin_prefetch(krow + (size_t)16 * Hn * En, 0, 0);
            v16h b0, b1;
#pragma unroll
            for (int i = 0; i < 16; ++i) {
                b0[i] = (_Float16)krow[i];
                b1[i] = (_Float16)krow[i + 32];
            }

            v8f c = {};
            c = __builtin_amdgcn_wmma_f32_16x16x32_f16(false, a0, false, b0, (short)0, c, false, false);
            c = __builtin_amdgcn_wmma_f32_16x16x32_f16(false, a1, false, b1, (short)0, c, false, false);

#pragma unroll
            for (int r = 0; r < 8; ++r) {
                const float sc = c[r];
                // C layout: M = r + 8*half, N = ln -> coalesced along s; NT store
                __builtin_nontemporal_store(sc,
                    attn_base + (size_t)(r + 8 * half) * Sn + s0 + ln);

                // per-lane online update (rescale only on an actual new max)
                if (sc > mpart[r]) {
                    lpart[r] *= __expf(mpart[r] - sc);
                    mpart[r] = sc;
                }
                lpart[r] += __expf(sc - mpart[r]);
            }
        }
    }

    // merge (m,l) across the 16 lanes of each half (once per kernel)
#pragma unroll
    for (int r = 0; r < 8; ++r) {
#pragma unroll
        for (int mask = 1; mask <= 8; mask <<= 1) {
            const float mo = __shfl_xor(mpart[r], mask);
            const float lo = __shfl_xor(lpart[r], mask);
            const float mn = fmaxf(mpart[r], mo);
            lpart[r] = lpart[r] * __expf(mpart[r] - mn) + lo * __expf(mo - mn);
            mpart[r] = mn;
        }
    }

    if (ln == 0) {
#pragma unroll
        for (int r = 0; r < 8; ++r) {
            const size_t idx = (size_t)(b * Hn + h) * Ln + l0 + r + 8 * half;
            mbuf[idx] = mpart[r];
            lbuf[idx] = lpart[r];
        }
    }
}

// ---------------------------------------------------------------------------
// Pass 2: p = exp(score - m) / l  (written back NT as the final attn map),
//         O = P @ V accumulated with WMMA. P tiles are transposed from the
//         C-fragment layout into the A-fragment layout via LDS.
// ---------------------------------------------------------------------------
__global__ __launch_bounds__(128) void attn_pv_kernel(
    const float* __restrict__ v, float* __restrict__ attn,
    const float* __restrict__ mbuf, const float* __restrict__ lbuf,
    float* __restrict__ out)
{
    __shared__ float lds[WAVES_PER_BLOCK][16][33]; // 16x32 p-tile per wave, padded

    const int lane = threadIdx.x & 31;
    const int wave = threadIdx.x >> 5;
    const int wid  = blockIdx.x * WAVES_PER_BLOCK + wave;

    const int lblocks = Ln / 16;
    const int lb  = wid % lblocks;
    int tmp       = wid / lblocks;
    const int h   = tmp % Hn;
    const int b   = tmp / Hn;
    const int l0  = lb * 16;

    const int ln   = lane & 15;
    const int half = lane >> 4;

    float mrow[8], linv[8];
#pragma unroll
    for (int r = 0; r < 8; ++r) {
        const size_t idx = (size_t)(b * Hn + h) * Ln + l0 + r + 8 * half;
        mrow[r] = mbuf[idx];
        linv[r] = 1.0f / lbuf[idx];
    }

    v8f acc[4];
#pragma unroll
    for (int et = 0; et < 4; ++et) acc[et] = (v8f){};

    float* attn_base = attn + ((size_t)(b * Hn + h) * Ln + l0) * Sn;
    float (*tile)[33] = lds[wave];

    for (int sc0 = 0; sc0 < Sn; sc0 += 32) {
        // normalize two 16-wide score tiles; stage transposed copy in LDS
#pragma unroll
        for (int t = 0; t < 2; ++t) {
            const int s0 = sc0 + t * 16;
#pragma unroll
            for (int r = 0; r < 8; ++r) {
                float* p_addr = attn_base + (size_t)(r + 8 * half) * Sn + s0 + ln;
                const float sc = __builtin_nontemporal_load(p_addr);
                const float p  = __expf(sc - mrow[r]) * linv[r];
                __builtin_nontemporal_store(p, p_addr);   // final attention map
                tile[r + 8 * half][t * 16 + ln] = p;
            }
        }
        // same-wave LDS store->load: DS ops are in-order; compiler inserts waits

        // A fragment of P (16 rows x 32 s) from LDS in the WMMA A layout
        v16h ap;
#pragma unroll
        for (int i = 0; i < 16; ++i) {
            const int kk = i + 16 * half;
            ap[i] = (_Float16)tile[ln][kk];
        }

        // B fragments of V (32 s x 16 e), one per 16-wide e tile
#pragma unroll
        for (int et = 0; et < 4; ++et) {
            v16h bv;
#pragma unroll
            for (int i = 0; i < 16; ++i) {
                const int s = sc0 + 16 * half + i;
                bv[i] = (_Float16)v[((size_t)(b * Sn + s) * Hn + h) * En + et * 16 + ln];
            }
            acc[et] = __builtin_amdgcn_wmma_f32_16x16x32_f16(false, ap, false, bv, (short)0,
                                                             acc[et], false, false);
        }
    }

    // out[b, l, h, e]: lanes map to consecutive e -> coalesced
#pragma unroll
    for (int et = 0; et < 4; ++et) {
#pragma unroll
        for (int r = 0; r < 8; ++r) {
            out[((size_t)(b * Ln + l0 + r + 8 * half) * Hn + h) * En + et * 16 + ln] =
                acc[et][r];
        }
    }
}

extern "C" void kernel_launch(void* const* d_in, const int* in_sizes, int n_in,
                              void* d_out, int out_size, void* d_ws, size_t ws_size,
                              hipStream_t stream) {
    const float* q = (const float*)d_in[0];
    const float* k = (const float*)d_in[1];
    const float* v = (const float*)d_in[2];

    float* out  = (float*)d_out;                       // [B,L,H,E]
    float* attn = out + (size_t)Bn * Ln * Hn * En;     // [B,H,L,S]

    float* mbuf = (float*)d_ws;                        // B*H*L floats
    float* lbuf = mbuf + (size_t)Bn * Hn * Ln;         // B*H*L floats

    const int total_waves = Bn * Hn * (Ln / 16);       // 8192
    const int blocks = total_waves / WAVES_PER_BLOCK;  // 2048

    attn_scores_kernel<<<blocks, 32 * WAVES_PER_BLOCK, 0, stream>>>(q, k, attn, mbuf, lbuf);
    attn_pv_kernel<<<blocks, 32 * WAVES_PER_BLOCK, 0, stream>>>(v, attn, mbuf, lbuf, out);
}
